// RSSM_19026705122008
// MI455X (gfx1250) — matrix operand order repaired and checked by reading, not compiled, not run on hardware
//
#include <hip/hip_runtime.h>
#include <hip/hip_bf16.h>

// ---------------------------------------------------------------------------
// RSSM rollout for MI455X (gfx1250).  bf16 WMMA 16x16x32, weights repacked
// once per call into B-fragment order so the whole weight set (~151 MB bf16)
// is L2-resident across the 64 sequential steps.
// ---------------------------------------------------------------------------

typedef __attribute__((ext_vector_type(16))) __bf16 v16bf;
typedef __attribute__((ext_vector_type(8)))  float  v8f;
typedef __attribute__((ext_vector_type(4)))  unsigned int uint4v;

#define WMMA_BF16(A, B, C) \
  __builtin_amdgcn_wmma_f32_16x16x32_bf16(false, (A), false, (B), (short)0, (C), false, false)

static constexpr int DETER = 4096, STOCH = 512, UNITS = 1024, ADIM = 2, EMBED = 4096;
static constexpr int BB = 16, TT = 64;
static constexpr int F_OUT = 11264;          // 6*512 + 2*4096
static constexpr int KCAT  = 544;            // 514 padded to 17 K-tiles of 32

__device__ __forceinline__ float sigf(float x) { return 1.f / (1.f + __expf(-x)); }

union FragU { v16bf v; uint4v q[2]; };

// A-fragment (16x32 bf16, row-major activations [16][row_stride])
__device__ __forceinline__ v16bf load_a_frag(const __bf16* base, int row_stride,
                                             int k0, int lane) {
  const int r = lane & 15, hi = lane >> 4;
  const __bf16* p = base + (size_t)r * row_stride + k0 + hi * 8;
  FragU u;
  u.q[0] = *(const uint4v*)(p);
  u.q[1] = *(const uint4v*)(p + 16);
  return u.v;
}

// B-fragment from packed tiles: tile = [32 lanes][16 bf16] contiguous (1 KB)
__device__ __forceinline__ v16bf load_b_frag(const __bf16* P, long tile, int lane) {
  return *(const v16bf*)(P + tile * 512 + (size_t)lane * 16);
}

// ---------------------------------------------------------------------------
// Weight repack: W[N][K] f32 row-major  ->  packed bf16 B-fragments.
// packed[((nt*KT + kt)*32 + lane)*16 + i] = W[nt*16 + (lane&15)][kt*32 + (lane>>4)*8 + (i&7) + (i>>3)*16]
// ---------------------------------------------------------------------------
__global__ void pack_b_kernel(const float* __restrict__ W, __bf16* __restrict__ P,
                              int K, int KT, long total) {
  long idx = (long)blockIdx.x * blockDim.x + threadIdx.x;
  if (idx >= total) return;
  int  i    = (int)(idx & 15);
  int  lane = (int)((idx >> 4) & 31);
  long tile = idx >> 9;
  int  kt   = (int)(tile % KT);
  int  nt   = (int)(tile / KT);
  int  n = nt * 16 + (lane & 15);
  int  k = kt * 32 + (lane >> 4) * 8 + (i & 7) + (i >> 3) * 16;
  float v = (k < K) ? W[(size_t)n * K + k] : 0.f;
  P[idx] = (__bf16)v;
}

__global__ void cvt_bf16_kernel(const float* __restrict__ src, __bf16* __restrict__ dst,
                                long n) {
  long i = (long)blockIdx.x * blockDim.x + threadIdx.x;
  if (i < n) dst[i] = (__bf16)src[i];
}

// deter0 = tanh(initial_deter) broadcast to [16,4096]; also bf16 copy for GEMM A
__global__ void init_state_kernel(const float* __restrict__ init_deter,
                                  float* __restrict__ deter_f32,
                                  __bf16* __restrict__ dnew_bf) {
  int idx = blockIdx.x * blockDim.x + threadIdx.x;     // 16*4096
  if (idx >= BB * DETER) return;
  int j = idx & (DETER - 1);
  float v = tanhf(init_deter[j]);
  deter_f32[idx] = v;
  dnew_bf[idx]   = (__bf16)v;
}

// stoch0 = mean part of stats(deter0 @ w_imgout.T + b)  -> 32 tiles of 16 cols
__global__ __launch_bounds__(32)
void init_stoch_kernel(const __bf16* __restrict__ dnew_bf,
                       const __bf16* __restrict__ P_imgout,
                       const float* __restrict__ b_imgout,
                       float* __restrict__ stoch_f32) {
  const int lane = threadIdx.x, jt = blockIdx.x;       // jt in [0,32)
  v8f acc = {};
  for (int kt = 0; kt < DETER / 32; ++kt) {
    v16bf a = load_a_frag(dnew_bf, DETER, kt * 32, lane);
    acc = WMMA_BF16(a, load_b_frag(P_imgout, (long)jt * (DETER / 32) + kt, lane), acc);
  }
  const int n = lane & 15, hi = lane >> 4;
#pragma unroll
  for (int v = 0; v < 8; ++v) {
    int m = v + 8 * hi, scol = jt * 16 + n;
    stoch_f32[m * STOCH + scol] = acc[v] + b_imgout[scol];
  }
}

// Per-step prep: mask by is_first, clip action, build concat([stoch,a]) bf16,
// masked deter in f32+bf16.
__global__ void prep_kernel(const float* __restrict__ deter_f32,
                            const float* __restrict__ stoch_f32,
                            const float* __restrict__ action,
                            const int* __restrict__ is_first, int t,
                            float* __restrict__ detm_f32,
                            __bf16* __restrict__ detm_bf,
                            __bf16* __restrict__ acat) {
  int idx = blockIdx.x * blockDim.x + threadIdx.x;
  const int ND = BB * DETER, NA = BB * KCAT;
  if (idx < ND) {
    int b = idx >> 12;
    float mask = is_first[b * TT + t] ? 0.f : 1.f;
    float v = deter_f32[idx] * mask;
    detm_f32[idx] = v;
    detm_bf[idx]  = (__bf16)v;
  } else if (idx < ND + NA) {
    int j = idx - ND, b = j / KCAT, c = j - b * KCAT;
    float mask = is_first[b * TT + t] ? 0.f : 1.f;
    float v = 0.f;
    if (c < STOCH) {
      v = stoch_f32[b * STOCH + c] * mask;
    } else if (c < STOCH + ADIM) {
      float a = action[((size_t)b * TT + t) * ADIM + (c - STOCH)];
      a = a * (1.f / fmaxf(fabsf(a), 1.f));         // ACTION_CLIP = 1
      v = a * mask;
    }
    acat[b * KCAT + c] = (__bf16)v;
  }
}

// x = concat(stoch_m, a_m) @ w_imgin.T + b_imgin   -> x_bf16 [16,1024]
__global__ __launch_bounds__(32)
void imgin_kernel(const __bf16* __restrict__ acat,
                  const __bf16* __restrict__ P_imgin,
                  const float* __restrict__ b_imgin,
                  __bf16* __restrict__ x_bf) {
  const int lane = threadIdx.x, jt = blockIdx.x;       // jt in [0,64)
  const int KT = KCAT / 32;                            // 17
  v8f acc = {};
  for (int kt = 0; kt < KT; ++kt) {
    v16bf a = load_a_frag(acat, KCAT, kt * 32, lane);
    acc = WMMA_BF16(a, load_b_frag(P_imgin, (long)jt * KT + kt, lane), acc);
  }
  const int n = lane & 15, hi = lane >> 4;
#pragma unroll
  for (int v = 0; v < 8; ++v) {
    int m = v + 8 * hi, col = jt * 16 + n;
    x_bf[m * UNITS + col] = (__bf16)(acc[v] + b_imgin[col]);
  }
}

// Fused GRU: 3 gi GEMVs (K=1024) + 3 gh GEMVs (K=4096) + gate math per tile.
// Writes d_new (f32 carry + bf16 A) and scatters d_new into out[...,1536] / [...,7168].
__global__ __launch_bounds__(32)
void gru_kernel(const __bf16* __restrict__ x_bf,
                const __bf16* __restrict__ detm_bf,
                const float* __restrict__ detm_f32,
                const __bf16* __restrict__ P_ih,
                const __bf16* __restrict__ P_hh,
                const float* __restrict__ b_ih,
                const float* __restrict__ b_hh,
                float* __restrict__ deter_f32,
                __bf16* __restrict__ dnew_bf,
                float* __restrict__ out, int t) {
  const int lane = threadIdx.x, jt = blockIdx.x;       // jt in [0,256)
  const int KT_I = UNITS / 32;                         // 32
  const int KT_H = DETER / 32;                         // 128
  const long ntR = jt, ntZ = 256 + jt, ntN = 512 + jt; // gate row-tiles
  v8f ir = {}, iz = {}, in = {}, hr = {}, hz = {}, hn = {};

  for (int kt = 0; kt < KT_I; ++kt) {
    v16bf a = load_a_frag(x_bf, UNITS, kt * 32, lane);
    ir = WMMA_BF16(a, load_b_frag(P_ih, ntR * KT_I + kt, lane), ir);
    iz = WMMA_BF16(a, load_b_frag(P_ih, ntZ * KT_I + kt, lane), iz);
    in = WMMA_BF16(a, load_b_frag(P_ih, ntN * KT_I + kt, lane), in);
  }
  for (int kt = 0; kt < KT_H; ++kt) {
    if (kt + 8 < KT_H)   // stream next w_hh tiles toward L0 (global_prefetch_b8)
      __builtin_prefetch(P_hh + (ntR * KT_H + kt + 8) * 512 + (size_t)lane * 16, 0, 0);
    v16bf a = load_a_frag(detm_bf, DETER, kt * 32, lane);
    hr = WMMA_BF16(a, load_b_frag(P_hh, ntR * KT_H + kt, lane), hr);
    hz = WMMA_BF16(a, load_b_frag(P_hh, ntZ * KT_H + kt, lane), hz);
    hn = WMMA_BF16(a, load_b_frag(P_hh, ntN * KT_H + kt, lane), hn);
  }

  const int n = lane & 15, hi = lane >> 4;
#pragma unroll
  for (int v = 0; v < 8; ++v) {
    int m = v + 8 * hi, col = jt * 16 + n;
    float gir = ir[v] + b_ih[col];
    float giz = iz[v] + b_ih[DETER + col];
    float gin = in[v] + b_ih[2 * DETER + col];
    float ghr = hr[v] + b_hh[col];
    float ghz = hz[v] + b_hh[DETER + col];
    float ghn = hn[v] + b_hh[2 * DETER + col];
    float r = sigf(gir + ghr);
    float z = sigf(giz + ghz);
    float nn = tanhf(gin + r * ghn);
    float h = detm_f32[m * DETER + col];
    float d = (1.f - z) * nn + z * h;
    deter_f32[m * DETER + col] = d;
    dnew_bf[m * DETER + col]   = (__bf16)d;
    size_t ob = ((size_t)m * TT + t) * F_OUT;
    out[ob + 1536 + col] = d;
    out[ob + 7168 + col] = d;
  }
}

// Fused heads: blocks 0..31 prior (img_out), blocks 32..63 posterior (obs_out).
// Each block produces one 16-col mean tile and its paired std tile.
__global__ __launch_bounds__(32)
void head_kernel(const __bf16* __restrict__ dnew_bf,
                 const __bf16* __restrict__ emb_bf,    // [B,T,4096] bf16
                 const __bf16* __restrict__ P_imgout,
                 const __bf16* __restrict__ P_obs,
                 const float* __restrict__ b_imgout,
                 const float* __restrict__ b_obs,
                 const float* __restrict__ eps_prior,
                 const float* __restrict__ eps_post,
                 float* __restrict__ stoch_f32,
                 float* __restrict__ out, int t) {
  const int lane = threadIdx.x;
  const int blk = blockIdx.x;
  const int KT_D = DETER / 32;                         // 128

  if (blk < 32) {                                      // ---- prior ----
    const int jt = blk;
    v8f am = {}, as = {};
    for (int kt = 0; kt < KT_D; ++kt) {
      v16bf a = load_a_frag(dnew_bf, DETER, kt * 32, lane);
      am = WMMA_BF16(a, load_b_frag(P_imgout, (long)jt * KT_D + kt, lane), am);
      as = WMMA_BF16(a, load_b_frag(P_imgout, (long)(jt + 32) * KT_D + kt, lane), as);
    }
    const int n = lane & 15, hi = lane >> 4;
#pragma unroll
    for (int v = 0; v < 8; ++v) {
      int m = v + 8 * hi, scol = jt * 16 + n;
      float mean = am[v] + b_imgout[scol];
      float sraw = as[v] + b_imgout[STOCH + scol];
      float std = fmaxf(2.f * sigf(0.5f * sraw) + 0.1f, 1e-4f);
      float ep = eps_prior[((size_t)t * BB + m) * STOCH + scol];
      size_t ob = ((size_t)m * TT + t) * F_OUT;
      out[ob + 5632 + scol] = mean + std * ep;         // pstoch
      out[ob + 6144 + scol] = mean;                    // pmean
      out[ob + 6656 + scol] = std;                     // pstd
    }
  } else {                                             // ---- posterior ----
    const int jt = blk - 32;
    const int KT_O = (DETER + EMBED) / 32;             // 256
    v8f am = {}, as = {};
    for (int kt = 0; kt < KT_D; ++kt) {                // d_new half of concat
      v16bf a = load_a_frag(dnew_bf, DETER, kt * 32, lane);
      am = WMMA_BF16(a, load_b_frag(P_obs, (long)jt * KT_O + kt, lane), am);
      as = WMMA_BF16(a, load_b_frag(P_obs, (long)(jt + 32) * KT_O + kt, lane), as);
    }
    const __bf16* emb_t = emb_bf + (size_t)t * EMBED;  // row stride T*EMBED
    for (int kt = 0; kt < KT_D; ++kt) {                // embed half of concat
      v16bf a = load_a_frag(emb_t, TT * EMBED, kt * 32, lane);
      am = WMMA_BF16(a, load_b_frag(P_obs, (long)jt * KT_O + 128 + kt, lane), am);
      as = WMMA_BF16(a, load_b_frag(P_obs, (long)(jt + 32) * KT_O + 128 + kt, lane), as);
    }
    const int n = lane & 15, hi = lane >> 4;
#pragma unroll
    for (int v = 0; v < 8; ++v) {
      int m = v + 8 * hi, scol = jt * 16 + n;
      float mean = am[v] + b_obs[scol];
      float sraw = as[v] + b_obs[STOCH + scol];
      float std = fmaxf(2.f * sigf(0.5f * sraw) + 0.1f, 1e-4f);
      float eo = eps_post[((size_t)t * BB + m) * STOCH + scol];
      float st = mean + std * eo;                      // ostoch
      size_t ob = ((size_t)m * TT + t) * F_OUT;
      out[ob + 0 + scol]    = st;
      out[ob + 512 + scol]  = mean;
      out[ob + 1024 + scol] = std;
      stoch_f32[m * STOCH + scol] = st;                // carry for next step
    }
  }
}

// ---------------------------------------------------------------------------
extern "C" void kernel_launch(void* const* d_in, const int* in_sizes, int n_in,
                              void* d_out, int out_size, void* d_ws, size_t ws_size,
                              hipStream_t stream) {
  const float* embed      = (const float*)d_in[0];
  const float* action     = (const float*)d_in[1];
  const int*   is_first   = (const int*)d_in[2];
  const float* init_deter = (const float*)d_in[3];
  const float* w_obs      = (const float*)d_in[4];
  const float* b_obs      = (const float*)d_in[5];
  const float* w_imgin    = (const float*)d_in[6];
  const float* b_imgin    = (const float*)d_in[7];
  const float* w_imgout   = (const float*)d_in[8];
  const float* b_imgout   = (const float*)d_in[9];
  const float* w_ih       = (const float*)d_in[10];
  const float* w_hh       = (const float*)d_in[11];
  const float* b_ih       = (const float*)d_in[12];
  const float* b_hh       = (const float*)d_in[13];
  const float* eps_prior  = (const float*)d_in[14];
  const float* eps_post   = (const float*)d_in[15];
  float* out = (float*)d_out;

  // ---- workspace carve (bf16 weight pack ~151 MB + small activation bufs) ----
  char* ws = (char*)d_ws;
  size_t off = 0;
  auto take = [&](size_t bytes) {
    void* p = ws + off;
    off = (off + bytes + 255) & ~(size_t)255;
    return p;
  };
  __bf16* P_imgin  = (__bf16*)take((size_t)UNITS * KCAT * 2);
  __bf16* P_ih     = (__bf16*)take((size_t)3 * DETER * UNITS * 2);
  __bf16* P_hh     = (__bf16*)take((size_t)3 * DETER * DETER * 2);
  __bf16* P_imgout = (__bf16*)take((size_t)UNITS * DETER * 2);
  __bf16* P_obs    = (__bf16*)take((size_t)UNITS * (DETER + EMBED) * 2);
  __bf16* emb_bf   = (__bf16*)take((size_t)BB * TT * EMBED * 2);
  float*  deter_f32 = (float*)take((size_t)BB * DETER * 4);
  float*  detm_f32  = (float*)take((size_t)BB * DETER * 4);
  __bf16* detm_bf   = (__bf16*)take((size_t)BB * DETER * 2);
  __bf16* dnew_bf   = (__bf16*)take((size_t)BB * DETER * 2);
  float*  stoch_f32 = (float*)take((size_t)BB * STOCH * 4);
  __bf16* acat      = (__bf16*)take((size_t)BB * KCAT * 2);
  __bf16* x_bf      = (__bf16*)take((size_t)BB * UNITS * 2);
  (void)ws_size; (void)n_in; (void)in_sizes; (void)out_size;

  auto pack = [&](const float* W, __bf16* P, int N, int K, int KT) {
    long total = (long)(N / 16) * KT * 512;
    pack_b_kernel<<<(int)((total + 255) / 256), 256, 0, stream>>>(W, P, K, KT, total);
  };
  // ---- one-time (per call) weight repack to bf16 B-fragments ----
  pack(w_imgin,  P_imgin,  UNITS,     514,           KCAT / 32);
  pack(w_ih,     P_ih,     3 * DETER, UNITS,         UNITS / 32);
  pack(w_hh,     P_hh,     3 * DETER, DETER,         DETER / 32);
  pack(w_imgout, P_imgout, UNITS,     DETER,         DETER / 32);
  pack(w_obs,    P_obs,    UNITS,     DETER + EMBED, (DETER + EMBED) / 32);
  {
    long n = (long)BB * TT * EMBED;
    cvt_bf16_kernel<<<(int)((n + 255) / 256), 256, 0, stream>>>(embed, emb_bf, n);
  }

  // ---- initial state ----
  init_state_kernel<<<(BB * DETER + 255) / 256, 256, 0, stream>>>(
      init_deter, deter_f32, dnew_bf);
  init_stoch_kernel<<<32, 32, 0, stream>>>(dnew_bf, P_imgout, b_imgout, stoch_f32);

  // ---- sequential rollout: 4 kernels per step on one stream ----
  const int prep_n = BB * DETER + BB * KCAT;
  for (int t = 0; t < TT; ++t) {
    prep_kernel<<<(prep_n + 255) / 256, 256, 0, stream>>>(
        deter_f32, stoch_f32, action, is_first, t, detm_f32, detm_bf, acat);
    imgin_kernel<<<UNITS / 16, 32, 0, stream>>>(acat, P_imgin, b_imgin, x_bf);
    gru_kernel<<<DETER / 16, 32, 0, stream>>>(
        x_bf, detm_bf, detm_f32, P_ih, P_hh, b_ih, b_hh,
        deter_f32, dnew_bf, out, t);
    head_kernel<<<64, 32, 0, stream>>>(
        dnew_bf, emb_bf, P_imgout, P_obs, b_imgout, b_obs,
        eps_prior, eps_post, stoch_f32, out, t);
  }
}
